// SpectralAlexNet_81913616269406
// MI455X (gfx1250) — matrix operand order repaired and checked by reading, not compiled, not run on hardware
//
#include <hip/hip_runtime.h>
#include <hip/hip_bf16.h>
#include <math.h>

// ---------------- CDNA5 WMMA types ----------------
typedef __attribute__((ext_vector_type(16))) __bf16 v16bf;
typedef __attribute__((ext_vector_type(8)))  float  v8f;
typedef __attribute__((ext_vector_type(4)))  unsigned int u32x4;
typedef __attribute__((ext_vector_type(8)))  int          i32x8;

#define R_LOG 3.8f
#define B_   32
#define D_   4096
#define V_   8192
#define C1_  256
#define K1_  32
#define K2_  16
#define KC   64     // K-chunk per LDS stage (2 WMMA k-steps of 32)

// RNE pack two f32 -> packed bf16x2
__device__ __forceinline__ unsigned pack_bf16(float a, float b) {
  unsigned ua = __float_as_uint(a), ub = __float_as_uint(b);
  ua += 0x7fffu + ((ua >> 16) & 1u);
  ub += 0x7fffu + ((ub >> 16) & 1u);
  return (ua >> 16) | (ub & 0xffff0000u);
}

// ---------------- TDM: 1-D row DMA global->LDS (CDNA5 ISA ch.8 D#) ----------------
__device__ __forceinline__ void tdm_load_row_f32(const float* gsrc, unsigned lds_off, int nelem) {
  unsigned long long ga = (unsigned long long)(const void*)gsrc;
  u32x4 g0;
  g0.x = 1u;                                   // count=1, user descriptor
  g0.y = lds_off;                              // D#.lds_addr
  g0.z = (unsigned)(ga & 0xffffffffu);         // global_addr[31:0]
  g0.w = (unsigned)((ga >> 32) & 0x01ffffffu)  // global_addr[56:32]
         | (2u << 30);                         // type=2 (image)
  i32x8 g1;
  g1[0] = (2 << 16);                           // workgroup_mask=0, data_size=2 (4B)
  g1[1] = (nelem & 0xffff) << 16;              // tensor_dim0[15:0] @ bits63:48
  g1[2] = ((nelem >> 16) & 0xffff) | (1 << 16);// tensor_dim0[31:16], tensor_dim1=1
  g1[3] = (nelem & 0xffff) << 16;              // tile_dim0 @ bits127:112
  g1[4] = 1;                                   // tile_dim1=1, tile_dim2=0
  g1[5] = nelem;                               // tensor_dim0_stride[31:0]
  g1[6] = 0;
  g1[7] = 0;
  asm volatile("tensor_load_to_lds %0, %1" :: "s"(g0), "s"(g1) : "memory");
}

__device__ __forceinline__ void wait_tensorcnt0() {
#if __has_builtin(__builtin_amdgcn_s_wait_tensorcnt)
  __builtin_amdgcn_s_wait_tensorcnt(0);
#else
  asm volatile("s_wait_tensorcnt 0x0" ::: "memory");
#endif
}

// ---------------- WMMA fragment loaders (per CDNA5 ISA §7.12.2) ----------------
// A (16x32 bf16): lanes 0-15 -> M=lane, K = {0..7,16..23}; lanes 16-31 -> K = {8..15,24..31}
__device__ __forceinline__ v16bf load_frag_A(const __bf16* lds, int ldk, int m, int khalf, int kstep) {
  union { unsigned int u[8]; v16bf v; } r;
  const __bf16* row = lds + m * ldk;
#pragma unroll
  for (int g = 0; g < 8; ++g) {
    int k = kstep * 32 + ((g < 4) ? 0 : 16) + khalf * 8 + 2 * (g & 3);
    r.u[g] = *(const unsigned int*)(row + k);
  }
  return r.v;
}
// B (32x16 bf16): lane -> N = lane&15; lanes 0-15 hold K = {0..7,16..23}, lanes 16-31 hold K = {8..15,24..31}
__device__ __forceinline__ v16bf load_frag_B(const __bf16* lds, int ldk, int n, int khalf, int kstep) {
  union { unsigned int u[8]; v16bf v; } r;
  const __bf16* row = lds + n * ldk;
#pragma unroll
  for (int g = 0; g < 8; ++g) {
    int k = kstep * 32 + ((g < 4) ? 0 : 16) + khalf * 8 + 2 * (g & 3);
    r.u[g] = *(const unsigned int*)(row + k);
  }
  return r.v;
}

// ---------------- x -> bf16 (packed rows) ----------------
__global__ void __launch_bounds__(256) k_convert_x(const float* __restrict__ x,
                                                   unsigned short* __restrict__ xbf) {
  int i = blockIdx.x * 256 + threadIdx.x;
  if (i < B_ * D_) {
    __bf16 h = (__bf16)x[i];
    xbf[i] = *(const unsigned short*)&h;
  }
}

// ---------------- GEMM 1: sims = |x @ conj(bank)^T|  (dual-GEMM re/im, WMMA bf16) ----
__global__ void __launch_bounds__(256) k_gemm_sims(const unsigned short* __restrict__ xbf,
                                                   const float* __restrict__ bank_re,
                                                   const float* __restrict__ bank_im,
                                                   float* __restrict__ sims) {
  __shared__ __bf16 sA[B_][KC];
  __shared__ __bf16 sBre[128][KC];
  __shared__ __bf16 sBim[128][KC];
  const int tid  = threadIdx.x;
  const int lane = tid & 31;
  const int wave = tid >> 5;           // 0..7
  const int vbase = blockIdx.x * 128;  // N tile over V
  const int nb = wave * 16;
  const int khalf = lane >> 4;
  v8f are0 = {}, are1 = {}, aim0 = {}, aim1 = {};

  for (int k0 = 0; k0 < D_; k0 += KC) {
    // stage A tile: 32x(KC/2) uints == 256 uint4, one per thread
    {
      const uint4* xrow = (const uint4*)xbf;
      int r = tid >> 3, c4 = tid & 7;
      ((uint4*)&sA[0][0])[tid] = xrow[(r * (D_ / 2) + k0 / 2) / 4 + c4];
    }
    // stage B tiles: float4 loads, RNE-packed bf16x4 stores
#pragma unroll
    for (int it = 0; it < 8; ++it) {
      int idx = tid + it * 256;             // 128 rows x 16 float4
      int r = idx >> 4, c = (idx & 15) << 2;
      size_t g = (size_t)(vbase + r) * D_ + k0 + c;
      float4 fre = *(const float4*)(bank_re + g);
      float4 fim = *(const float4*)(bank_im + g);
      unsigned long long ure = (unsigned long long)pack_bf16(fre.x, fre.y)
                             | ((unsigned long long)pack_bf16(fre.z, fre.w) << 32);
      unsigned long long uim = (unsigned long long)pack_bf16(fim.x, fim.y)
                             | ((unsigned long long)pack_bf16(fim.z, fim.w) << 32);
      *(unsigned long long*)&sBre[r][c] = ure;
      *(unsigned long long*)&sBim[r][c] = uim;
    }
    // prefetch next chunk (global_prefetch_b8)
    if (k0 + KC < D_) {
      size_t g = (size_t)(vbase + (tid >> 1)) * D_ + k0 + KC + (tid & 1) * 32;
      __builtin_prefetch(&bank_re[g], 0, 1);
      __builtin_prefetch(&bank_im[g], 0, 1);
    }
    __syncthreads();
#pragma unroll
    for (int ks = 0; ks < 2; ++ks) {
      v16bf a0  = load_frag_A(&sA[0][0],  KC, (lane & 15),      khalf, ks);
      v16bf a1  = load_frag_A(&sA[0][0],  KC, 16 + (lane & 15), khalf, ks);
      v16bf bre = load_frag_B(&sBre[0][0], KC, nb + (lane & 15), khalf, ks);
      v16bf bim = load_frag_B(&sBim[0][0], KC, nb + (lane & 15), khalf, ks);
      are0 = __builtin_amdgcn_wmma_f32_16x16x32_bf16(false, a0, false, bre, (short)0, are0, false, false);
      are1 = __builtin_amdgcn_wmma_f32_16x16x32_bf16(false, a1, false, bre, (short)0, are1, false, false);
      aim0 = __builtin_amdgcn_wmma_f32_16x16x32_bf16(false, a0, false, bim, (short)0, aim0, false, false);
      aim1 = __builtin_amdgcn_wmma_f32_16x16x32_bf16(false, a1, false, bim, (short)0, aim1, false, false);
    }
    __syncthreads();
  }
  // epilogue: magnitude
  const int ncol = vbase + nb + (lane & 15);
  const int moff = khalf * 8;
#pragma unroll
  for (int r = 0; r < 8; ++r) {
    int m0 = r + moff;
    float re = are0[r], im = aim0[r];
    sims[m0 * V_ + ncol] = sqrtf(re * re + im * im);
    float re1 = are1[r], im1 = aim1[r];
    sims[(16 + m0) * V_ + ncol] = sqrtf(re1 * re1 + im1 * im1);
  }
}

// ---------------- softmax over V per row, emit bf16 attn ----------------
__global__ void __launch_bounds__(256) k_softmax(const float* __restrict__ sims,
                                                 unsigned short* __restrict__ attnbf) {
  __shared__ float red[256];
  const int b = blockIdx.x, tid = threadIdx.x;
  float mx = -1e30f;
  for (int i = tid; i < V_; i += 256) mx = fmaxf(mx, sims[b * V_ + i]);
  red[tid] = mx; __syncthreads();
  for (int s = 128; s > 0; s >>= 1) { if (tid < s) red[tid] = fmaxf(red[tid], red[tid + s]); __syncthreads(); }
  mx = red[0]; __syncthreads();
  float sum = 0.f;
  for (int i = tid; i < V_; i += 256) sum += expf(sims[b * V_ + i] - mx);
  red[tid] = sum; __syncthreads();
  for (int s = 128; s > 0; s >>= 1) { if (tid < s) red[tid] += red[tid + s]; __syncthreads(); }
  const float inv = 1.0f / red[0];
  for (int i = tid; i < V_; i += 256) {
    __bf16 h = (__bf16)(expf(sims[b * V_ + i] - mx) * inv);
    attnbf[b * V_ + i] = *(const unsigned short*)&h;
  }
}

// ---------------- GEMM 2: bank_out = attn @ bank (re & im) ----------------
__global__ void __launch_bounds__(256) k_gemm_bankout(const unsigned short* __restrict__ attnbf,
                                                      const float* __restrict__ bank_re,
                                                      const float* __restrict__ bank_im,
                                                      float* __restrict__ bo_re,
                                                      float* __restrict__ bo_im) {
  __shared__ __bf16 sA[B_][KC];
  __shared__ __bf16 sBre[128][KC];
  __shared__ __bf16 sBim[128][KC];
  const int tid  = threadIdx.x;
  const int lane = tid & 31;
  const int wave = tid >> 5;
  const int dbase = blockIdx.x * 128;  // N tile over D
  const int nb = wave * 16;
  const int khalf = lane >> 4;
  v8f are0 = {}, are1 = {}, aim0 = {}, aim1 = {};

  for (int k0 = 0; k0 < V_; k0 += KC) {
    {
      const uint4* arow = (const uint4*)attnbf;
      int r = tid >> 3, c4 = tid & 7;
      ((uint4*)&sA[0][0])[tid] = arow[(r * (V_ / 2) + k0 / 2) / 4 + c4];
    }
    // B[k=v][n=d]: load two consecutive v-rows as float4, pack K-pairs, transpose to LDS
#pragma unroll
    for (int it = 0; it < 4; ++it) {
      int idx = tid + it * 256;             // (KC/2) x 32 tasks
      int k2 = (idx >> 5) << 1, n4 = (idx & 31) << 2;
      const float* p0re = bank_re + (size_t)(k0 + k2) * D_ + dbase + n4;
      const float* p0im = bank_im + (size_t)(k0 + k2) * D_ + dbase + n4;
      float4 r0 = *(const float4*)p0re;
      float4 r1 = *(const float4*)(p0re + D_);
      float4 q0 = *(const float4*)p0im;
      float4 q1 = *(const float4*)(p0im + D_);
      *(unsigned*)&sBre[n4 + 0][k2] = pack_bf16(r0.x, r1.x);
      *(unsigned*)&sBre[n4 + 1][k2] = pack_bf16(r0.y, r1.y);
      *(unsigned*)&sBre[n4 + 2][k2] = pack_bf16(r0.z, r1.z);
      *(unsigned*)&sBre[n4 + 3][k2] = pack_bf16(r0.w, r1.w);
      *(unsigned*)&sBim[n4 + 0][k2] = pack_bf16(q0.x, q1.x);
      *(unsigned*)&sBim[n4 + 1][k2] = pack_bf16(q0.y, q1.y);
      *(unsigned*)&sBim[n4 + 2][k2] = pack_bf16(q0.z, q1.z);
      *(unsigned*)&sBim[n4 + 3][k2] = pack_bf16(q0.w, q1.w);
    }
    if (k0 + KC < V_) {
      size_t g = (size_t)(k0 + KC + (tid >> 2)) * D_ + dbase + (tid & 3) * 32;
      __builtin_prefetch(&bank_re[g], 0, 1);
      __builtin_prefetch(&bank_im[g], 0, 1);
    }
    __syncthreads();
#pragma unroll
    for (int ks = 0; ks < 2; ++ks) {
      v16bf a0  = load_frag_A(&sA[0][0],  KC, (lane & 15),      khalf, ks);
      v16bf a1  = load_frag_A(&sA[0][0],  KC, 16 + (lane & 15), khalf, ks);
      v16bf bre = load_frag_B(&sBre[0][0], KC, nb + (lane & 15), khalf, ks);
      v16bf bim = load_frag_B(&sBim[0][0], KC, nb + (lane & 15), khalf, ks);
      are0 = __builtin_amdgcn_wmma_f32_16x16x32_bf16(false, a0, false, bre, (short)0, are0, false, false);
      are1 = __builtin_amdgcn_wmma_f32_16x16x32_bf16(false, a1, false, bre, (short)0, are1, false, false);
      aim0 = __builtin_amdgcn_wmma_f32_16x16x32_bf16(false, a0, false, bim, (short)0, aim0, false, false);
      aim1 = __builtin_amdgcn_wmma_f32_16x16x32_bf16(false, a1, false, bim, (short)0, aim1, false, false);
    }
    __syncthreads();
  }
  const int ncol = dbase + nb + (lane & 15);
  const int moff = khalf * 8;
#pragma unroll
  for (int r = 0; r < 8; ++r) {
    int m0 = r + moff;
    bo_re[m0 * D_ + ncol] = are0[r];
    bo_im[m0 * D_ + ncol] = aim0[r];
    bo_re[(16 + m0) * D_ + ncol] = are1[r];
    bo_im[(16 + m0) * D_ + ncol] = aim1[r];
  }
}

// ---------------- logistic coupling on real x (per-row min/max of |x|) ----------------
__global__ void __launch_bounds__(256) k_logistic_rows(const float* __restrict__ x,
                                                       float* __restrict__ xc) {
  __shared__ float red[256];
  const int b = blockIdx.x, tid = threadIdx.x;
  float mn = 1e30f, mx = -1e30f;
  for (int i = tid; i < D_; i += 256) {
    float m = fabsf(x[b * D_ + i]);
    mn = fminf(mn, m); mx = fmaxf(mx, m);
  }
  red[tid] = mn; __syncthreads();
  for (int s = 128; s > 0; s >>= 1) { if (tid < s) red[tid] = fminf(red[tid], red[tid + s]); __syncthreads(); }
  mn = red[0]; __syncthreads();
  red[tid] = mx; __syncthreads();
  for (int s = 128; s > 0; s >>= 1) { if (tid < s) red[tid] = fmaxf(red[tid], red[tid + s]); __syncthreads(); }
  mx = red[0];
  const float inv = 1.0f / (mx - mn + 1e-10f);
  for (int i = tid; i < D_; i += 256) {
    float v = x[b * D_ + i];
    float xn = (fabsf(v) - mn) * inv;
#pragma unroll
    for (int it = 0; it < 5; ++it) xn = R_LOG * xn * (1.0f - xn);
    xc[b * D_ + i] = copysignf(xn * (mx - mn) + mn, v);
  }
}

// ---------------- radix-2 DIT FFT, 4096-pt, one row per block; TDM row staging ------
__global__ void __launch_bounds__(256) k_fft4096(const float* __restrict__ in_re,
                                                 const float* __restrict__ in_im, int has_im,
                                                 float* __restrict__ out_re,
                                                 float* __restrict__ out_im) {
  __shared__ float sr[D_];
  __shared__ float si[D_];
  const int b = blockIdx.x, tid = threadIdx.x;
  // Tensor Data Mover: DMA the whole row into LDS (natural order), wave 0 issues.
  if (tid == 0) {
    tdm_load_row_f32(in_re + (size_t)b * D_, (unsigned)(size_t)(void*)sr, D_);
    if (has_im)
      tdm_load_row_f32(in_im + (size_t)b * D_, (unsigned)(size_t)(void*)si, D_);
    wait_tensorcnt0();
  }
  __syncthreads();
  // in-place bit-reversal (involution) + zero imag if absent
  for (int i = tid; i < D_; i += 256) {
    int j = (int)(__brev((unsigned)i) >> 20);
    if (j > i) {
      float t = sr[i]; sr[i] = sr[j]; sr[j] = t;
      if (has_im) { float u = si[i]; si[i] = si[j]; si[j] = u; }
    }
  }
  if (!has_im)
    for (int i = tid; i < D_; i += 256) si[i] = 0.f;
  __syncthreads();
  for (int s = 1; s <= 12; ++s) {
    const int m = 1 << s, half = m >> 1;
    for (int idx = tid; idx < D_ / 2; idx += 256) {
      int grp = idx / half, pos = idx - grp * half;
      int i = grp * m + pos, j = i + half;
      float ang = -6.28318530717958647692f * (float)pos / (float)m;
      float wi_, wr_; __sincosf(ang, &wi_, &wr_);
      float tr = wr_ * sr[j] - wi_ * si[j];
      float ti = wr_ * si[j] + wi_ * sr[j];
      sr[j] = sr[i] - tr; si[j] = si[i] - ti;
      sr[i] += tr;        si[i] += ti;
    }
    __syncthreads();
  }
  for (int i = tid; i < D_; i += 256) {
    out_re[b * D_ + i] = sr[i];
    out_im[b * D_ + i] = si[i];
  }
}

// ---------------- small DFT of k1 (256 x 32) ----------------
__global__ void __launch_bounds__(256) k_dft_k1(const float* __restrict__ k1_re,
                                                const float* __restrict__ k1_im,
                                                float* __restrict__ kf_re,
                                                float* __restrict__ kf_im) {
  int gid = blockIdx.x * 256 + threadIdx.x;
  if (gid >= C1_ * K1_) return;
  int o = gid >> 5, t = gid & 31;
  float re = 0.f, im = 0.f;
  for (int k = 0; k < K1_; ++k) {
    float th = 6.28318530717958647692f * (float)(k * t) / (float)K1_;
    float s, c; __sincosf(th, &s, &c);
    float a = k1_re[o * K1_ + k], bb = k1_im[o * K1_ + k];
    re += a * c + bb * s;        // (a+ib) e^{-i th}
    im += bb * c - a * s;
  }
  kf_re[gid] = re; kf_im[gid] = im;
}

// ---------------- w[m] = (1/L^2) sum_l sin(a*l) e^{+2pi i l m / L} ----------------
__global__ void __launch_bounds__(256) k_compute_w(const float* __restrict__ alpha,
                                                   float* __restrict__ w_re,
                                                   float* __restrict__ w_im) {
  int m = blockIdx.x * 256 + threadIdx.x;
  if (m >= D_) return;
  const float a = alpha[0];
  float re = 0.f, im = 0.f;
  for (int l = 0; l < D_; ++l) {
    float sl = sinf(a * (float)l);
    int t = (l * m) & (D_ - 1);
    float th = 6.28318530717958647692f * (float)t / (float)D_;
    float s, c; __sincosf(th, &s, &c);
    re += sl * c; im += sl * s;
  }
  const float sc = 1.0f / ((float)D_ * (float)D_);
  w_re[m] = re * sc; w_im[m] = im * sc;
}

// ---------------- feats[b,o] = sum_m (xf*w)[b,m] * ki[o,m] ----------------
__global__ void __launch_bounds__(256) k_feats(const float* __restrict__ xf_re,
                                               const float* __restrict__ xf_im,
                                               const float* __restrict__ w_re,
                                               const float* __restrict__ w_im,
                                               const float* __restrict__ kf_re,
                                               const float* __restrict__ kf_im,
                                               float* __restrict__ fe_re,
                                               float* __restrict__ fe_im) {
  __shared__ float r1[256], r2[256];
  const int b = blockIdx.x / C1_, o = blockIdx.x % C1_, tid = threadIdx.x;
  float ar = 0.f, ai = 0.f;
  for (int m = tid; m < D_; m += 256) {
    float pos = ((float)m + 0.5f) * ((float)K1_ / (float)D_) - 0.5f;
    pos = fminf(fmaxf(pos, 0.f), (float)(K1_ - 1));
    int i0 = (int)floorf(pos);
    int i1 = (i0 + 1 < K1_) ? i0 + 1 : K1_ - 1;
    float wg = pos - (float)i0;
    float r0 = kf_re[o * K1_ + i0], q0 = kf_im[o * K1_ + i0];
    float r1v = kf_re[o * K1_ + i1], q1 = kf_im[o * K1_ + i1];
    float km = (1.f - wg) * sqrtf(r0 * r0 + q0 * q0) + wg * sqrtf(r1v * r1v + q1 * q1);
    float kp = (1.f - wg) * atan2f(q0, r0) + wg * atan2f(q1, r1v);
    float ks, kc; __sincosf(kp, &ks, &kc);
    float kr = km * kc, kii = km * ks;
    float xr = xf_re[b * D_ + m], xi = xf_im[b * D_ + m];
    float gr = xr * w_re[m] - xi * w_im[m];
    float gi = xr * w_im[m] + xi * w_re[m];
    ar += gr * kr - gi * kii;
    ai += gr * kii + gi * kr;
  }
  r1[tid] = ar; r2[tid] = ai; __syncthreads();
  for (int s = 128; s > 0; s >>= 1) {
    if (tid < s) { r1[tid] += r1[tid + s]; r2[tid] += r2[tid + s]; }
    __syncthreads();
  }
  if (tid == 0) { fe_re[b * C1_ + o] = r1[0]; fe_im[b * C1_ + o] = r2[0]; }
}

__device__ __forceinline__ void hamilton4(const float* q, const float* p, float* d) {
  d[0] = q[0]*p[0] - q[1]*p[1] - q[2]*p[2] - q[3]*p[3];
  d[1] = q[0]*p[1] + q[1]*p[0] + q[2]*p[3] - q[3]*p[2];
  d[2] = q[0]*p[2] - q[1]*p[3] + q[2]*p[0] + q[3]*p[1];
  d[3] = q[0]*p[3] + q[1]*p[2] - q[2]*p[1] + q[3]*p[0];
}

// ---------------- quaternion pipeline + analytic layer-3 scalar ----------------
__global__ void k_quat(const float* __restrict__ fe_re, const float* __restrict__ fe_im,
                       const float* __restrict__ wq, const float* __restrict__ bq,
                       const float* __restrict__ qL, const float* __restrict__ qR,
                       const float* __restrict__ k2_re, const float* __restrict__ k2_im,
                       float* __restrict__ coupled_out, float* __restrict__ sb) {
  const int b = threadIdx.x;
  if (b >= B_) return;
  float q[4];
#pragma unroll
  for (int i = 0; i < 4; ++i) {
    float s = bq[i];
    for (int j = 0; j < C1_; ++j) s += fe_re[b * C1_ + j] * wq[i * (2 * C1_) + j];
    for (int j = 0; j < C1_; ++j) s += fe_im[b * C1_ + j] * wq[i * (2 * C1_) + C1_ + j];
    q[i] = s;
  }
  float n = sqrtf(q[0]*q[0] + q[1]*q[1] + q[2]*q[2] + q[3]*q[3]) + 1e-10f;
#pragma unroll
  for (int i = 0; i < 4; ++i) q[i] /= n;
  float ql[4] = {qL[0], qL[1], qL[2], qL[3]};
  float qr[4] = {qR[0], qR[1], qR[2], qR[3]};
  float h1[4], ev[4];
  hamilton4(ql, q, h1);
  hamilton4(h1, qr, ev);
  float c[4];
#pragma unroll
  for (int i = 0; i < 4; ++i) {
    float y = (ev[i] + 1.f) * 0.5f;
#pragma unroll
    for (int it = 0; it < 3; ++it) y = R_LOG * y * (1.f - y);
    c[i] = 2.f * y - 1.f;
  }
  n = sqrtf(c[0]*c[0] + c[1]*c[1] + c[2]*c[2] + c[3]*c[3]) + 1e-10f;
#pragma unroll
  for (int i = 0; i < 4; ++i) c[i] /= n;
#pragma unroll
  for (int i = 0; i < 4; ++i) coupled_out[b * 4 + i] = c[i];
  // layer-3 analytic: input constant along L -> FFT is DC delta -> scalar * sin(a2*l)
  float mn = 1e30f, mx = -1e30f;
#pragma unroll
  for (int i = 0; i < 4; ++i) { float m = fabsf(c[i]); mn = fminf(mn, m); mx = fmaxf(mx, m); }
  float sre = 0.f, sim = 0.f;
#pragma unroll
  for (int i = 0; i < 4; ++i) {
    float xn = (fabsf(c[i]) - mn) / (mx - mn + 1e-10f);
#pragma unroll
    for (int it = 0; it < 3; ++it) xn = R_LOG * xn * (1.f - xn);
    float val = copysignf(xn * (mx - mn) + mn, c[i]);
    float kdr = 0.f, kdi = 0.f;
    for (int k = 0; k < K2_; ++k) { kdr += k2_re[i * K2_ + k]; kdi += k2_im[i * K2_ + k]; }
    sre += val * kdr; sim += val * kdi;
  }
  sb[b * 2] = sre; sb[b * 2 + 1] = sim;
}

// ---------------- interp[b,l] = s_b * sin(a2*l) ----------------
__global__ void __launch_bounds__(256) k_interp_fill(const float* __restrict__ sb,
                                                     const float* __restrict__ alpha2,
                                                     float* __restrict__ ir,
                                                     float* __restrict__ ii) {
  int gid = blockIdx.x * 256 + threadIdx.x;
  if (gid >= B_ * D_) return;
  int b = gid / D_, l = gid % D_;
  float s = sinf(alpha2[0] * (float)l);
  ir[gid] = sb[b * 2] * s;
  ii[gid] = sb[b * 2 + 1] * s;
}

// ---------------- per-row spectral stats ----------------
__global__ void __launch_bounds__(256) k_stats(const float* s1r, const float* s1i,
                                               const float* s2r, const float* s2i,
                                               const float* s3r, const float* s3i,
                                               float* __restrict__ rowstats) {
  __shared__ float red[256];
  const int b = blockIdx.x, tid = threadIdx.x;
  float loc[12];
#pragma unroll
  for (int q = 0; q < 12; ++q) loc[q] = 0.f;
  for (int i = tid; i < D_; i += 256) {
    int g = b * D_ + i;
    float ar = s1r[g], ai = s1i[g], br = s2r[g], bi = s2i[g], cr = s3r[g], ci = s3i[g];
    loc[0] += ar * br + ai * bi;  loc[1] += ai * br - ar * bi;   // <s1, conj(s2)>
    loc[2] += ar * cr + ai * ci;  loc[3] += ai * cr - ar * ci;   // <s1, conj(s3)>
    loc[4] += br * cr + bi * ci;  loc[5] += bi * cr - br * ci;   // <s2, conj(s3)>
    loc[6] += ar * ar + ai * ai;  loc[7] += br * br + bi * bi;  loc[8] += cr * cr + ci * ci;
    float dr = ar - br, di = ai - bi; loc[9]  += dr * dr + di * di;
    dr = ar - cr; di = ai - ci;       loc[10] += dr * dr + di * di;
    dr = br - cr; di = bi - ci;       loc[11] += dr * dr + di * di;
  }
#pragma unroll
  for (int q = 0; q < 12; ++q) {
    red[tid] = loc[q]; __syncthreads();
    for (int s = 128; s > 0; s >>= 1) { if (tid < s) red[tid] += red[tid + s]; __syncthreads(); }
    if (tid == 0) rowstats[b * 12 + q] = red[0];
    __syncthreads();
  }
}

__global__ void k_finalize(const float* __restrict__ rowstats, float* __restrict__ out) {
  __shared__ float red[6][B_];
  const int b = threadIdx.x;
  if (b < B_) {
    const float* r = rowstats + b * 12;
    float e1 = r[6], e2 = r[7], e3 = r[8];
    red[0][b] = (r[0]*r[0] + r[1]*r[1]) / (e1 * e2 + 1e-10f);
    red[1][b] = (r[2]*r[2] + r[3]*r[3]) / (e1 * e3 + 1e-10f);
    red[2][b] = (r[4]*r[4] + r[5]*r[5]) / (e2 * e3 + 1e-10f);
    red[3][b] = sqrtf(r[9]);
    red[4][b] = sqrtf(r[10]);
    red[5][b] = sqrtf(r[11]);
  }
  __syncthreads();
  if (b == 0) {
    float s[6] = {0, 0, 0, 0, 0, 0};
    for (int i = 0; i < B_; ++i)
      for (int q = 0; q < 6; ++q) s[q] += red[q][i];
    for (int q = 0; q < 6; ++q) s[q] /= (float)B_;
    out[2 * B_ * D_]     = (s[0] + s[1] + s[2]) / 3.0f;           // resonance
    out[2 * B_ * D_ + 1] = 1.0f / (1.0f + s[3] + s[4] + s[5]);    // coherence
  }
}

// ---------------- final = 0.3*bank_out + 0.3*x + 0.4*interp (interleaved complex) ----
__global__ void __launch_bounds__(256) k_final(const float* __restrict__ bo_re,
                                               const float* __restrict__ bo_im,
                                               const float* __restrict__ x,
                                               const float* __restrict__ ir,
                                               const float* __restrict__ ii,
                                               float* __restrict__ out) {
  int gid = blockIdx.x * 256 + threadIdx.x;
  if (gid >= B_ * D_) return;
  out[2 * gid]     = 0.3f * bo_re[gid] + 0.3f * x[gid] + 0.4f * ir[gid];
  out[2 * gid + 1] = 0.3f * bo_im[gid] + 0.4f * ii[gid];
}

// ================================================================================
extern "C" void kernel_launch(void* const* d_in, const int* in_sizes, int n_in,
                              void* d_out, int out_size, void* d_ws, size_t ws_size,
                              hipStream_t stream) {
  (void)in_sizes; (void)n_in; (void)out_size; (void)ws_size;
  const float* x       = (const float*)d_in[0];
  const float* bank_re = (const float*)d_in[1];
  const float* bank_im = (const float*)d_in[2];
  const float* k1_re   = (const float*)d_in[3];
  const float* k1_im   = (const float*)d_in[4];
  const float* alpha1  = (const float*)d_in[5];
  const float* wq      = (const float*)d_in[6];
  const float* bq      = (const float*)d_in[7];
  const float* qL      = (const float*)d_in[8];
  const float* qR      = (const float*)d_in[9];
  const float* k2_re   = (const float*)d_in[10];
  const float* k2_im   = (const float*)d_in[11];
  const float* alpha2  = (const float*)d_in[12];
  float* out = (float*)d_out;

  char* wsb = (char*)d_ws;
  size_t off = 0;
  auto alloc = [&](size_t bytes) -> void* {
    void* p = wsb + off;
    off += (bytes + 255) & ~(size_t)255;
    return p;
  };
  unsigned short* xbf    = (unsigned short*)alloc((size_t)B_ * D_ * 2);
  unsigned short* attnbf = (unsigned short*)alloc((size_t)B_ * V_ * 2);
  float* sims  = (float*)alloc((size_t)B_ * V_ * 4);
  float* bo_re = (float*)alloc((size_t)B_ * D_ * 4);
  float* bo_im = (float*)alloc((size_t)B_ * D_ * 4);
  float* xc    = (float*)alloc((size_t)B_ * D_ * 4);
  float* xf_re = (float*)alloc((size_t)B_ * D_ * 4);
  float* xf_im = (float*)alloc((size_t)B_ * D_ * 4);
  float* s1r   = (float*)alloc((size_t)B_ * D_ * 4);
  float* s1i   = (float*)alloc((size_t)B_ * D_ * 4);
  float* s2r   = (float*)alloc((size_t)B_ * D_ * 4);
  float* s2i   = (float*)alloc((size_t)B_ * D_ * 4);
  float* s3r   = (float*)alloc((size_t)B_ * D_ * 4);
  float* s3i   = (float*)alloc((size_t)B_ * D_ * 4);
  float* ir    = (float*)alloc((size_t)B_ * D_ * 4);
  float* ii    = (float*)alloc((size_t)B_ * D_ * 4);
  float* w_re  = (float*)alloc((size_t)D_ * 4);
  float* w_im  = (float*)alloc((size_t)D_ * 4);
  float* kf_re = (float*)alloc((size_t)C1_ * K1_ * 4);
  float* kf_im = (float*)alloc((size_t)C1_ * K1_ * 4);
  float* fe_re = (float*)alloc((size_t)B_ * C1_ * 4);
  float* fe_im = (float*)alloc((size_t)B_ * C1_ * 4);
  float* sb    = (float*)alloc((size_t)B_ * 2 * 4);
  float* rowst = (float*)alloc((size_t)B_ * 12 * 4);

  // Layer 1: bank attention (WMMA GEMMs + softmax + WMMA GEMMs)
  k_convert_x<<<(B_ * D_ + 255) / 256, 256, 0, stream>>>(x, xbf);
  k_gemm_sims<<<V_ / 128, 256, 0, stream>>>(xbf, bank_re, bank_im, sims);
  k_softmax<<<B_, 256, 0, stream>>>(sims, attnbf);
  k_gemm_bankout<<<D_ / 128, 256, 0, stream>>>(attnbf, bank_re, bank_im, bo_re, bo_im);

  // Layer 2: spectral conv (analytic probe/ifft folding into w[m])
  k_logistic_rows<<<B_, 256, 0, stream>>>(x, xc);
  k_fft4096<<<B_, 256, 0, stream>>>(xc, nullptr, 0, xf_re, xf_im);
  k_fft4096<<<B_, 256, 0, stream>>>(x, nullptr, 0, s2r, s2i);       // s2 = fft(x)
  k_dft_k1<<<(C1_ * K1_ + 255) / 256, 256, 0, stream>>>(k1_re, k1_im, kf_re, kf_im);
  k_compute_w<<<D_ / 256, 256, 0, stream>>>(alpha1, w_re, w_im);
  k_feats<<<B_ * C1_, 256, 0, stream>>>(xf_re, xf_im, w_re, w_im, kf_re, kf_im, fe_re, fe_im);

  // Quaternion evolution + coupling; analytic layer 3
  k_quat<<<1, B_, 0, stream>>>(fe_re, fe_im, wq, bq, qL, qR, k2_re, k2_im,
                               out + 2 * B_ * D_ + 2, sb);
  k_interp_fill<<<(B_ * D_ + 255) / 256, 256, 0, stream>>>(sb, alpha2, ir, ii);

  // Spectral comparison
  k_fft4096<<<B_, 256, 0, stream>>>(bo_re, bo_im, 1, s1r, s1i);
  k_fft4096<<<B_, 256, 0, stream>>>(ir, ii, 1, s3r, s3i);
  k_stats<<<B_, 256, 0, stream>>>(s1r, s1i, s2r, s2i, s3r, s3i, rowst);
  k_finalize<<<1, B_, 0, stream>>>(rowst, out);
  k_final<<<(B_ * D_ + 255) / 256, 256, 0, stream>>>(bo_re, bo_im, x, ir, ii, out);
}
